// ArcMarginHead_10368051052865
// MI455X (gfx1250) — matrix operand order repaired and checked by reading, not compiled, not run on hardware
//
#include <hip/hip_runtime.h>
#include <math.h>

// ---------------------------------------------------------------------------
// ArcMargin head for MI455X (gfx1250, wave32, WMMA)
//   cos = l2norm(E) @ l2norm(W)^T ; ArcFace margin epilogue ; *64
// B=2048, D=512, C=100000.
// Path: fp32 -> normalized f16 in d_ws (W_f16 = 102 MB fits the 192 MB L2),
// then f16 WMMA GEMM with f32 accumulators and a fused epilogue.
// Workspace use: (2048+100000)*512*2 bytes ~= 104.5 MB.
// ---------------------------------------------------------------------------

typedef __attribute__((ext_vector_type(16))) _Float16 v16h;
typedef __attribute__((ext_vector_type(8)))  _Float16 v8h;
typedef __attribute__((ext_vector_type(8)))  float    v8f;
typedef __attribute__((ext_vector_type(4)))  float    v4f;

#define B_ROWS 2048
#define D_DIM  512
#define C_COLS 100000

// ArcFace constants (margin = 0.5, scale = 64)
#define ARC_COS_M  0.8775825618903728f   //  cos(0.5)
#define ARC_SIN_M  0.4794255386042030f   //  sin(0.5)
#define ARC_TH    -0.8775825618903728f   //  cos(pi - 0.5)
#define ARC_MM     0.2397127693021015f   //  sin(pi - 0.5) * 0.5
#define ARC_SCALE  64.0f

__device__ __forceinline__ float wave32_reduce_sum(float v) {
    // butterfly across the 32 lanes of a CDNA5 wave
    #pragma unroll
    for (int off = 16; off > 0; off >>= 1)
        v += __shfl_xor(v, off, 32);
    return v;
}

// One wave per row: load 512 f32, L2-normalize, emit 512 f16.
__global__ __launch_bounds__(256)
void normalize_rows_f32_to_f16(const float* __restrict__ src,
                               _Float16* __restrict__ dst,
                               int nrows) {
    const int lane = threadIdx.x & 31;
    const int wave = threadIdx.x >> 5;
    const int row  = blockIdx.x * 8 + wave;
    if (row >= nrows) return;

    const float* p = src + (size_t)row * D_DIM + lane * 16;
    v4f x0 = *(const v4f*)(p + 0);
    v4f x1 = *(const v4f*)(p + 4);
    v4f x2 = *(const v4f*)(p + 8);
    v4f x3 = *(const v4f*)(p + 12);

    float ss = 0.f;
    #pragma unroll
    for (int i = 0; i < 4; ++i)
        ss += x0[i]*x0[i] + x1[i]*x1[i] + x2[i]*x2[i] + x3[i]*x3[i];
    ss = wave32_reduce_sum(ss);

    const float r = 1.0f / fmaxf(sqrtf(ss), 1e-12f);

    v8h h0, h1;
    #pragma unroll
    for (int i = 0; i < 4; ++i) {
        h0[i]     = (_Float16)(x0[i] * r);
        h0[i + 4] = (_Float16)(x1[i] * r);
        h1[i]     = (_Float16)(x2[i] * r);
        h1[i + 4] = (_Float16)(x3[i] * r);
    }
    _Float16* q = dst + (size_t)row * D_DIM + lane * 16;
    *(v8h*)(q)     = h0;
    *(v8h*)(q + 8) = h1;
}

// Block: 256 threads = 8 waves, tile 128(M) x 128(N).
// Wave: 32(M) x 64(N) = 2x4 WMMA 16x16 tiles, K-loop over D=512 in steps of 32.
__global__ __launch_bounds__(256)
void arc_margin_wmma(const _Float16* __restrict__ A,   // [B, D] normalized f16
                     const _Float16* __restrict__ W,   // [C, D] normalized f16
                     const int* __restrict__ labels,   // [B]
                     float* __restrict__ out)          // [B, C] f32
{
    const int lane  = threadIdx.x & 31;
    const int wave  = threadIdx.x >> 5;
    const int l15   = lane & 15;
    const int lhalf = lane >> 4;          // 0 or 1 (half-wave)

    const int waveM = wave & 3;           // 4 waves along M
    const int waveN = wave >> 2;          // 2 waves along N

    const int mBase = blockIdx.x * 128 + waveM * 32;
    const int nBase = blockIdx.y * 128 + waveN * 64;

    // A-fragment row pointers (A layout: lanes 0-15 and 16-31 both hold M=l15,
    // but different K sub-ranges -> handled via lhalf offsets below).
    const _Float16* aRow0 = A + (size_t)(mBase + 0  + l15) * D_DIM;
    const _Float16* aRow1 = A + (size_t)(mBase + 16 + l15) * D_DIM;

    // B-fragment row pointers: lane stripes N; each lane reads K-contiguous
    // data from W row n. Clamp the C-tail for loads; stores are guarded.
    const _Float16* wRow[4];
    int nCol[4];
    #pragma unroll
    for (int j = 0; j < 4; ++j) {
        int n = nBase + j * 16 + l15;
        nCol[j] = n;
        int nc = (n < C_COLS) ? n : (C_COLS - 1);
        wRow[j] = W + (size_t)nc * D_DIM;
    }

    v8f acc[2][4];
    #pragma unroll
    for (int t = 0; t < 2; ++t)
        #pragma unroll
        for (int j = 0; j < 4; ++j)
            acc[t][j] = (v8f){0.f, 0.f, 0.f, 0.f, 0.f, 0.f, 0.f, 0.f};

    const int aOff = lhalf * 16;   // A: half-wave 0 -> K 0..15,       half 1 -> K 16..31
    const int bOff = lhalf * 16;   // B: half-wave 0 -> K 0..15,       half 1 -> K 16..31

    for (int k0 = 0; k0 < D_DIM; k0 += 32) {
        v16h afrag[2], bfrag[4];
        {
            v8h c1 = *(const v8h*)(aRow0 + k0 + aOff);
            v8h c2 = *(const v8h*)(aRow0 + k0 + 8 + aOff);
            afrag[0] = __builtin_shufflevector(c1, c2, 0,1,2,3,4,5,6,7,8,9,10,11,12,13,14,15);
            v8h d1 = *(const v8h*)(aRow1 + k0 + aOff);
            v8h d2 = *(const v8h*)(aRow1 + k0 + 8 + aOff);
            afrag[1] = __builtin_shufflevector(d1, d2, 0,1,2,3,4,5,6,7,8,9,10,11,12,13,14,15);
        }
        #pragma unroll
        for (int j = 0; j < 4; ++j) {
            const _Float16* p = wRow[j] + k0 + bOff;
            v8h c1 = *(const v8h*)(p);
            v8h c2 = *(const v8h*)(p + 8);
            bfrag[j] = __builtin_shufflevector(c1, c2, 0,1,2,3,4,5,6,7,8,9,10,11,12,13,14,15);
        }
        #pragma unroll
        for (int t = 0; t < 2; ++t)
            #pragma unroll
            for (int j = 0; j < 4; ++j)
                acc[t][j] = __builtin_amdgcn_wmma_f32_16x16x32_f16(
                    /*neg_a=*/false, afrag[t],
                    /*neg_b=*/false, bfrag[j],
                    /*c_mod=*/(short)0, acc[t][j],
                    /*reuse_a=*/false, /*reuse_b=*/false);
    }

    // Fused ArcFace epilogue + store.
    // C/D layout: VGPR i, lanes 0-15 -> (M = i,   N = l15)
    //             VGPR i, lanes 16-31 -> (M = i+8, N = l15)
    #pragma unroll
    for (int t = 0; t < 2; ++t) {
        const int mSub = mBase + t * 16 + lhalf * 8;
        int lab[8];
        #pragma unroll
        for (int i = 0; i < 8; ++i)
            lab[i] = labels[mSub + i];

        #pragma unroll
        for (int j = 0; j < 4; ++j) {
            const int n = nCol[j];
            #pragma unroll
            for (int i = 0; i < 8; ++i) {
                float c   = acc[t][j][i];
                float s   = sqrtf(fmaxf(0.f, 1.f - c * c));
                float phi = fmaf(c, ARC_COS_M, -(s * ARC_SIN_M));
                if (!(c > ARC_TH)) phi = c - ARC_MM;
                float logit = (lab[i] == n) ? phi : c;
                if (n < C_COLS)
                    out[(size_t)(mSub + i) * C_COLS + n] = ARC_SCALE * logit;
            }
        }
    }
}

extern "C" void kernel_launch(void* const* d_in, const int* in_sizes, int n_in,
                              void* d_out, int out_size, void* d_ws, size_t ws_size,
                              hipStream_t stream) {
    (void)in_sizes; (void)n_in; (void)out_size; (void)ws_size;

    const float* emb    = (const float*)d_in[0];   // [2048, 512]
    const int*   labels = (const int*)d_in[1];     // [2048]
    const float* weight = (const float*)d_in[2];   // [100000, 512]
    float*       out    = (float*)d_out;           // [2048, 100000]

    _Float16* A16 = (_Float16*)d_ws;                       // 2 MB
    _Float16* W16 = A16 + (size_t)B_ROWS * D_DIM;          // 102.4 MB

    normalize_rows_f32_to_f16<<<B_ROWS / 8, 256, 0, stream>>>(emb, A16, B_ROWS);
    normalize_rows_f32_to_f16<<<(C_COLS + 7) / 8, 256, 0, stream>>>(weight, W16, C_COLS);

    // M-tiles on x (16), N-tiles on y (782): consecutively launched blocks share
    // the same 128-row slice of W16, which the 192 MB L2 serves after one HBM read.
    dim3 grid(B_ROWS / 128, (C_COLS + 127) / 128);
    arc_margin_wmma<<<grid, 256, 0, stream>>>(A16, W16, labels, out);
}
